// MultiHeadAttention_76510547411319
// MI455X (gfx1250) — matrix operand order repaired and checked by reading, not compiled, not run on hardware
//
#include <hip/hip_runtime.h>
#include <hip/hip_bf16.h>

#define B_ 4
#define T_ 2048
#define C_ 1024
#define H_ 16
#define D_ 64

typedef __attribute__((ext_vector_type(16))) __bf16 v16bf;
typedef __attribute__((ext_vector_type(8)))  float  v8f;

union BF16x16 {
    v16bf v;
    unsigned short u[16];
    uint4 q[2];
};

__device__ __forceinline__ unsigned short f32_to_bf16(float f) {
    unsigned int x = __float_as_uint(f);
    x = (x + 0x7FFFu + ((x >> 16) & 1u)) >> 16;   // round-to-nearest-even
    return (unsigned short)x;
}

__device__ __forceinline__ float bf16_to_f32(unsigned short h) {
    return __uint_as_float(((unsigned int)h) << 16);
}

// ---------------------------------------------------------------------------
// f32 -> bf16 conversion, 4 elements / thread
// ---------------------------------------------------------------------------
__global__ __launch_bounds__(256)
void cvt_bf16x4_kernel(const float4* __restrict__ src, uint2* __restrict__ dst, int n4) {
    int i = blockIdx.x * 256 + threadIdx.x;
    if (i < n4) {
        float4 f = src[i];
        union { unsigned short u[4]; uint2 d; } p;
        p.u[0] = f32_to_bf16(f.x);
        p.u[1] = f32_to_bf16(f.y);
        p.u[2] = f32_to_bf16(f.z);
        p.u[3] = f32_to_bf16(f.w);
        dst[i] = p.d;
    }
}

// ---------------------------------------------------------------------------
// Async global->LDS staging of one (A 128x32, W 64x32) bf16 tile.
// 3 x global_load_async_to_lds_b128 per wave, tracked by ASYNCcnt.
// Instruction offset applies to BOTH lds and global address (ISA 08 §4.4).
// ---------------------------------------------------------------------------
__device__ __forceinline__ void stage_tile_async(
    const unsigned short* __restrict__ A,
    const unsigned short* __restrict__ W,
    unsigned int ldsA, unsigned int ldsW,     // LDS byte addresses of buffers
    int bm0, int bn0, int K, int k0, int tid)
{
    {   // A tile: thread -> row tid>>1, 16-elem half (tid&1); two 16B asyncs
        const int row = tid >> 1;
        const int h16 = (tid & 1) * 16;
        unsigned long long g =
            (unsigned long long)(const void*)(A + (size_t)(bm0 + row) * K + k0 + h16);
        unsigned int l = ldsA + (unsigned int)(row * 80 + h16 * 2);
        asm volatile("global_load_async_to_lds_b128 %0, %1, off"
                     :: "v"(l), "v"(g) : "memory");
        asm volatile("global_load_async_to_lds_b128 %0, %1, off offset:16"
                     :: "v"(l), "v"(g) : "memory");
    }
    {   // W tile: thread -> row tid>>2, 8-elem quarter; one 16B async
        const int row = tid >> 2;
        const int q8  = (tid & 3) * 8;
        unsigned long long g =
            (unsigned long long)(const void*)(W + (size_t)(bn0 + row) * K + k0 + q8);
        unsigned int l = ldsW + (unsigned int)(row * 80 + q8 * 2);
        asm volatile("global_load_async_to_lds_b128 %0, %1, off"
                     :: "v"(l), "v"(g) : "memory");
    }
}

// ---------------------------------------------------------------------------
// GEMM: out[M,N] = A[M,K](bf16) @ W[N,K]^T(bf16) + bias
// EPI==0: write f32 row-major to outF
// EPI==1: qkv GEMM: q,k -> bf16 [B,H,T,D]; v -> bf16 [B,H,D,T] (transposed,
//         packed b128 store: the 8 accum rows of a fragment are contiguous t)
// Tiles: workgroup 128(M) x 64(N), 8 waves in 4x2, each wave 32x32 via 2x2
// WMMA. Double-buffered LDS filled by async global->LDS loads.
// ---------------------------------------------------------------------------
template<int EPI>
__global__ __launch_bounds__(256)
void gemm_bias_kernel(const unsigned short* __restrict__ A,
                      const unsigned short* __restrict__ W,
                      const float*          __restrict__ bias,
                      float*                __restrict__ outF,
                      unsigned short*       __restrict__ qb,
                      unsigned short*       __restrict__ kb,
                      unsigned short*       __restrict__ vb,
                      int M, int N, int K)
{
    // padded row stride 40 halfwords (80B): 16B-aligned rows, bank-conflict free
    __shared__ __align__(16) unsigned short a_sh[2][128 * 40];
    __shared__ __align__(16) unsigned short w_sh[2][64 * 40];

    const int tid  = threadIdx.x;
    const int lane = tid & 31;
    const int wave = tid >> 5;
    const int g    = lane >> 4;     // lane group (0: lanes 0-15, 1: lanes 16-31)
    const int nl   = lane & 15;
    const int wm   = wave >> 1;     // 0..3
    const int wn   = wave & 1;      // 0..1
    const int bm0  = blockIdx.y * 128;
    const int bn0  = blockIdx.x * 64;

    // LDS byte addresses (generic pointer low 32 bits == LDS offset)
    const unsigned int ldsA0 = (unsigned int)(size_t)(void*)&a_sh[0][0];
    const unsigned int ldsA1 = (unsigned int)(size_t)(void*)&a_sh[1][0];
    const unsigned int ldsW0 = (unsigned int)(size_t)(void*)&w_sh[0][0];
    const unsigned int ldsW1 = (unsigned int)(size_t)(void*)&w_sh[1][0];

    const v8f vzero = {0.f, 0.f, 0.f, 0.f, 0.f, 0.f, 0.f, 0.f};
    v8f acc[2][2] = {{vzero, vzero}, {vzero, vzero}};

    const int nIter = K / 32;
    stage_tile_async(A, W, ldsA0, ldsW0, bm0, bn0, K, 0, tid);

    for (int i = 0; i < nIter; ++i) {
        const int cur = i & 1;
        __syncthreads();   // all waves done reading buffer cur^1 (prev compute)
        if (i + 1 < nIter) {
            stage_tile_async(A, W, cur ? ldsA0 : ldsA1, cur ? ldsW0 : ldsW1,
                             bm0, bn0, K, (i + 1) * 32, tid);
            asm volatile("s_wait_asynccnt 0x3" ::: "memory"); // tile i landed
        } else {
            asm volatile("s_wait_asynccnt 0x0" ::: "memory");
        }
        __syncthreads();   // every wave's async data for buffer cur is visible

        BF16x16 af[2], bfr[2];
#pragma unroll
        for (int fm = 0; fm < 2; ++fm) {
            // A 16x32 bf16 layout: lane m=nl; k = g*16 + e
            const unsigned short* p = &a_sh[cur][(wm * 32 + fm * 16 + nl) * 40];
            af[fm].q[0] = *(const uint4*)(p + g * 16);
            af[fm].q[1] = *(const uint4*)(p + g * 16 + 8);
        }
#pragma unroll
        for (int fn = 0; fn < 2; ++fn) {
            // B 32x16 bf16 layout: lane n=nl; k = g*16 + e  (read W transposed)
            const unsigned short* p = &w_sh[cur][(wn * 32 + fn * 16 + nl) * 40] + g * 16;
            bfr[fn].q[0] = ((const uint4*)p)[0];
            bfr[fn].q[1] = ((const uint4*)p)[1];
        }
#pragma unroll
        for (int fm = 0; fm < 2; ++fm)
#pragma unroll
            for (int fn = 0; fn < 2; ++fn)
                acc[fm][fn] = __builtin_amdgcn_wmma_f32_16x16x32_bf16(
                    false, af[fm].v, false, bfr[fn].v, (short)0, acc[fm][fn], false, false);
    }

    // epilogue: C/D layout -> lane n=nl, VGPR r -> m = g*8 + r
#pragma unroll
    for (int fm = 0; fm < 2; ++fm) {
#pragma unroll
        for (int fn = 0; fn < 2; ++fn) {
            const int col = bn0 + wn * 32 + fn * 16 + nl;
            const float bv = bias[col];
            if (EPI == 0) {
#pragma unroll
                for (int r = 0; r < 8; ++r) {
                    const int row = bm0 + wm * 32 + fm * 16 + g * 8 + r;
                    outF[(size_t)row * N + col] = acc[fm][fn][r] + bv;
                }
            } else {
                const int which = col >> 10;        // uniform per fragment
                const int cin   = col & 1023;
                const int h     = cin >> 6;
                const int d     = cin & 63;
                const int row0  = bm0 + wm * 32 + fm * 16 + g * 8;
                const int b     = row0 >> 11;
                const int t8    = row0 & 2047;      // first of 8 consecutive t
                if (which == 2) {
                    // V^T [B,H,D,T]: 8 consecutive t per lane -> one b128 store
                    union { unsigned short u[8]; uint4 q; } pk;
#pragma unroll
                    for (int r = 0; r < 8; ++r)
                        pk.u[r] = f32_to_bf16(acc[fm][fn][r] + bv);
                    *(uint4*)(vb + ((size_t)(b * H_ + h) * D_ + d) * T_ + t8) = pk.q;
                } else {
                    unsigned short* dst = (which == 0) ? qb : kb;
#pragma unroll
                    for (int r = 0; r < 8; ++r) {
                        const size_t idx =
                            (((size_t)(b * H_ + h) * T_) + (t8 + r)) * D_ + d;
                        dst[idx] = f32_to_bf16(acc[fm][fn][r] + bv);
                    }
                }
            }
        }
    }
}

// ---------------------------------------------------------------------------
// RoPE in place on q and k (bf16 [B,H,T,D]); each thread owns pair (d, d+32)
// ---------------------------------------------------------------------------
__global__ __launch_bounds__(256)
void rope_kernel(unsigned short* __restrict__ q, unsigned short* __restrict__ k,
                 const float* __restrict__ sincos /* [2,T,D] */) {
    const int idx = blockIdx.x * 256 + threadIdx.x;   // B*H*T*32 threads total
    const int d   = idx & 31;
    const int t   = (idx >> 5) & (T_ - 1);
    const int bh  = idx >> 16;                        // 32*2048 = 1<<16
    const float s_lo = sincos[t * D_ + d];
    const float s_hi = sincos[t * D_ + d + 32];
    const float c_lo = sincos[T_ * D_ + t * D_ + d];
    const float c_hi = sincos[T_ * D_ + t * D_ + d + 32];
    const size_t base = ((size_t)bh * T_ + t) * D_;
    {
        float lo = bf16_to_f32(q[base + d]);
        float hi = bf16_to_f32(q[base + d + 32]);
        q[base + d]      = f32_to_bf16(lo * c_lo - hi * s_lo);
        q[base + d + 32] = f32_to_bf16(hi * c_hi + lo * s_hi);
    }
    {
        float lo = bf16_to_f32(k[base + d]);
        float hi = bf16_to_f32(k[base + d + 32]);
        k[base + d]      = f32_to_bf16(lo * c_lo - hi * s_lo);
        k[base + d + 32] = f32_to_bf16(hi * c_hi + lo * s_hi);
    }
}

// ---------------------------------------------------------------------------
// Fused attention: one workgroup per (b, h, 16 query rows).
// QK^T (WMMA) -> 16x2048 f32 scores in LDS -> softmax (wave32 shuffles)
// -> stream attn to global once -> P@V (WMMA; probs from LDS, V^T read as
//    two b128 per fragment) -> ds_add_f32 reduction -> ctx bf16 [B,T,H*D]
// ---------------------------------------------------------------------------
#define SC_STRIDE 2052              // 2048 + 4 pad (keeps 16B align, avoids conflicts)
#define ATTN_SMEM_BYTES (size_t)((16 * SC_STRIDE + 16 * 64) * sizeof(float))

__global__ __launch_bounds__(256)
void attn_kernel(const unsigned short* __restrict__ q,
                 const unsigned short* __restrict__ k,
                 const unsigned short* __restrict__ vT,  // [B,H,D,T]
                 float*                __restrict__ attn, // [B,H,T,T]
                 unsigned short*       __restrict__ ctx)  // [B,T,H*D]
{
    extern __shared__ char smem[];
    float* sc   = (float*)smem;                                    // [16][SC_STRIDE]
    float* o_sh = (float*)(smem + 16 * SC_STRIDE * sizeof(float)); // [16][64]

    const int tid  = threadIdx.x;
    const int lane = tid & 31;
    const int wave = tid >> 5;
    const int g    = lane >> 4;
    const int nl   = lane & 15;

    const int qt = blockIdx.x & 127;
    const int h  = (blockIdx.x >> 7) & 15;
    const int b  = blockIdx.x >> 11;
    const int t0 = qt * 16;
    const size_t bh = (size_t)b * H_ + h;
    const size_t hb = bh * T_ * D_;

    // Q A-fragments (16 x 64 -> two 16x32 bf16 fragments), straight from global
    BF16x16 qf0, qf1;
    {
        const unsigned short* qp = q + hb + (size_t)(t0 + nl) * D_;
        qf0.q[0] = *(const uint4*)(qp + g * 16);
        qf0.q[1] = *(const uint4*)(qp + g * 16 + 8);
        qf1.q[0] = *(const uint4*)(qp + 32 + g * 16);
        qf1.q[1] = *(const uint4*)(qp + 32 + g * 16 + 8);
    }

    for (int i = tid; i < 16 * 64; i += 256) o_sh[i] = 0.f;

    const v8f vzero = {0.f, 0.f, 0.f, 0.f, 0.f, 0.f, 0.f, 0.f};

    // ---- scores: wave w handles key tiles w, w+8, ... ----
    for (int ct = wave; ct < 128; ct += 8) {
        const int key0 = ct * 16;
        // B frags: element (kk, n) = K[key0+n][kk]; lane n=nl, kk = g*16+e
        const unsigned short* kp = k + hb + (size_t)(key0 + nl) * D_;
        BF16x16 kf0, kf1;
        kf0.q[0] = ((const uint4*)(kp + g * 16))[0];
        kf0.q[1] = ((const uint4*)(kp + g * 16))[1];
        kf1.q[0] = ((const uint4*)(kp + 32 + g * 16))[0];
        kf1.q[1] = ((const uint4*)(kp + 32 + g * 16))[1];
        v8f acc = vzero;
        acc = __builtin_amdgcn_wmma_f32_16x16x32_bf16(false, qf0.v, false, kf0.v, (short)0, acc, false, false);
        acc = __builtin_amdgcn_wmma_f32_16x16x32_bf16(false, qf1.v, false, kf1.v, (short)0, acc, false, false);
#pragma unroll
        for (int r = 0; r < 8; ++r)
            sc[(g * 8 + r) * SC_STRIDE + key0 + nl] = acc[r] * 0.125f;  // * D^-0.5
    }
    __syncthreads();

    // ---- softmax: wave w owns rows 2w, 2w+1 ----
#pragma unroll
    for (int rr = 0; rr < 2; ++rr) {
        const int row = wave * 2 + rr;
        float* srow = sc + row * SC_STRIDE;
        float mx = -3.40282e38f;
        for (int c = lane; c < 2048; c += 32) mx = fmaxf(mx, srow[c]);
#pragma unroll
        for (int off = 16; off > 0; off >>= 1) mx = fmaxf(mx, __shfl_xor(mx, off, 32));
        float sum = 0.f;
        for (int c = lane; c < 2048; c += 32) {
            float e = __expf(srow[c] - mx);
            srow[c] = e;
            sum += e;
        }
#pragma unroll
        for (int off = 16; off > 0; off >>= 1) sum += __shfl_xor(sum, off, 32);
        const float inv = 1.f / sum;
        float* arow = attn + (bh * T_ + (size_t)(t0 + row)) * T_;
        for (int c = lane * 4; c < 2048; c += 128) {
            float4 p = *(float4*)(srow + c);
            p.x *= inv; p.y *= inv; p.z *= inv; p.w *= inv;
            *(float4*)(srow + c) = p;       // normalized probs stay in LDS for P@V
            *(float4*)(arow + c) = p;       // single streaming write of attn
        }
    }
    __syncthreads();

    // ---- P@V: wave w covers keys [w*256, w*256+256), all 4 d-tiles ----
    v8f oacc[4] = {vzero, vzero, vzero, vzero};
    const unsigned short* vbase = vT + bh * D_ * T_;
    for (int kt = 0; kt < 8; ++kt) {
        const int key0 = (wave * 8 + kt) * 32;
        BF16x16 pf;
        const float* prow = sc + nl * SC_STRIDE + key0;
#pragma unroll
        for (int e = 0; e < 16; ++e) {
            const int kk = g * 16 + e;
            pf.u[e] = f32_to_bf16(prow[kk]);
        }
#pragma unroll
        for (int fn = 0; fn < 4; ++fn) {
            // B frag: element (kk=key, n=d) = V^T[d][key]; per lane 16
            // contiguous keys -> two b128 loads
            const unsigned short* vp =
                vbase + (size_t)(fn * 16 + nl) * T_ + key0 + g * 16;
            BF16x16 vf;
            vf.q[0] = ((const uint4*)vp)[0];
            vf.q[1] = ((const uint4*)vp)[1];
            oacc[fn] = __builtin_amdgcn_wmma_f32_16x16x32_bf16(
                false, pf.v, false, vf.v, (short)0, oacc[fn], false, false);
        }
    }
#pragma unroll
    for (int fn = 0; fn < 4; ++fn)
#pragma unroll
        for (int r = 0; r < 8; ++r)
            atomicAdd(&o_sh[(g * 8 + r) * 64 + fn * 16 + nl], oacc[fn][r]);
    __syncthreads();

    for (int i = tid; i < 16 * 64; i += 256) {
        const int m = i >> 6;
        const int d = i & 63;
        ctx[(size_t)(b * T_ + t0 + m) * C_ + h * D_ + d] = f32_to_bf16(o_sh[i]);
    }
}

// ---------------------------------------------------------------------------
extern "C" void kernel_launch(void* const* d_in, const int* in_sizes, int n_in,
                              void* d_out, int out_size, void* d_ws, size_t ws_size,
                              hipStream_t stream) {
    const float* x     = (const float*)d_in[0];
    const float* rope  = (const float*)d_in[1];
    const float* Wqkv  = (const float*)d_in[2];
    const float* bqkv  = (const float*)d_in[3];
    const float* Wproj = (const float*)d_in[4];
    const float* bproj = (const float*)d_in[5];

    float* out  = (float*)d_out;
    float* attn = out + (size_t)B_ * T_ * C_;          // outputs concatenated: out, attn

    // workspace layout (bytes)
    char* ws = (char*)d_ws;
    unsigned short* xb    = (unsigned short*)(ws + 0);          // 16,777,216
    unsigned short* wqkvb = (unsigned short*)(ws + 16777216);   //  6,291,456
    unsigned short* wprojb= (unsigned short*)(ws + 23068672);   //  2,097,152
    unsigned short* qb    = (unsigned short*)(ws + 25165824);   // 16,777,216  [B,H,T,D]
    unsigned short* kb    = (unsigned short*)(ws + 41943040);   // 16,777,216  [B,H,T,D]
    unsigned short* vTb   = (unsigned short*)(ws + 58720256);   // 16,777,216  [B,H,D,T]
    unsigned short* ctxb  = (unsigned short*)(ws + 75497472);   // 16,777,216  [B,T,C]

    const int nX  = B_ * T_ * C_;        // 8,388,608
    const int nWq = 3 * C_ * C_;         // 3,145,728
    const int nWp = C_ * C_;             // 1,048,576

    cvt_bf16x4_kernel<<<(nX / 4 + 255) / 256, 256, 0, stream>>>((const float4*)x, (uint2*)xb, nX / 4);
    cvt_bf16x4_kernel<<<(nWq / 4 + 255) / 256, 256, 0, stream>>>((const float4*)Wqkv, (uint2*)wqkvb, nWq / 4);
    cvt_bf16x4_kernel<<<(nWp / 4 + 255) / 256, 256, 0, stream>>>((const float4*)Wproj, (uint2*)wprojb, nWp / 4);

    // qkv = x @ Wqkv^T + b; q,k -> [B,H,T,D], v -> [B,H,D,T]
    gemm_bias_kernel<1><<<dim3(3 * C_ / 64, B_ * T_ / 128), 256, 0, stream>>>(
        xb, wqkvb, bqkv, nullptr, qb, kb, vTb, B_ * T_, 3 * C_, C_);

    rope_kernel<<<(B_ * H_ * T_ * 32) / 256, 256, 0, stream>>>(qb, kb, rope);

    hipFuncSetAttribute(reinterpret_cast<const void*>(attn_kernel),
                        hipFuncAttributeMaxDynamicSharedMemorySize, (int)ATTN_SMEM_BYTES);
    attn_kernel<<<B_ * H_ * (T_ / 16), 256, ATTN_SMEM_BYTES, stream>>>(qb, kb, vTb, attn, ctxb);

    // out = ctx @ Wproj^T + b
    gemm_bias_kernel<0><<<dim3(C_ / 64, B_ * T_ / 128), 256, 0, stream>>>(
        ctxb, wprojb, bproj, out, nullptr, nullptr, nullptr, B_ * T_, C_, C_);
}
